// EncoderSARNN_19052474925181
// MI455X (gfx1250) — matrix operand order repaired and checked by reading, not compiled, not run on hardware
//
#include <hip/hip_runtime.h>
#include <hip/hip_bf16.h>

#define T_ 512
#define B_ 64
#define IDIM_ 256
#define HDIM_ 512
#define NSTACK_ 2
#define SSZ_ 32
#define SDIM_ 64
#define NWG_ 8

typedef __attribute__((ext_vector_type(16))) __bf16 v16bf;
typedef __attribute__((ext_vector_type(8)))  float  v8f;

// ---- workspace byte offsets (all 256B aligned) ----
#define OFF_WE   0u          // bf16 [512][256]
#define OFF_WH   262144u     // bf16 [512][512]
#define OFF_WR   786432u     // bf16 [1024][256]
#define OFF_WS   1310720u    // bf16 [128][512]
#define OFF_WA   1441792u    // bf16 [16][512] (rows 6..15 zero-padded)
#define OFF_PRE  1458176u    // bf16 [T*B][512]
#define OFF_HID  35012608u   // bf16 [2][64][512] ping-pong
#define OFF_STK  35143680u   // f32  [2][64][2][32][64] ping-pong
#define OFF_CNT  37240832u   // u32 barrier counter

__device__ __forceinline__ v8f wmma_bf16(v16bf a, v16bf b, v8f c) {
  return __builtin_amdgcn_wmma_f32_16x16x32_bf16(false, a, false, b,
                                                 (short)0, c, false, false);
}

// A fragment 16x32 bf16, row-major source with stride lda (elements).
// Lane L: M = m0 + (L&15); lanes 0-15 cover K {0..7,16..23}, lanes 16-31 {8..15,24..31}.
__device__ __forceinline__ v16bf load_a_bf16(const __bf16* A, int lda, int m0,
                                             int k0, int lane) {
  const __bf16* p = A + (size_t)(m0 + (lane & 15)) * lda + k0;
  int kb = (lane >> 4) << 3;
  v16bf a;
#pragma unroll
  for (int i = 0; i < 8; ++i) {
    int k = ((i & 4) << 2) + kb + ((i & 3) << 1);
    a[2 * i]     = p[k];
    a[2 * i + 1] = p[k + 1];
  }
  return a;
}

// Same A layout but converting from fp32 source on the fly.
__device__ __forceinline__ v16bf load_a_f32(const float* A, int lda, int m0,
                                            int k0, int lane) {
  const float* p = A + (size_t)(m0 + (lane & 15)) * lda + k0;
  int kb = (lane >> 4) << 3;
  v16bf a;
#pragma unroll
  for (int i = 0; i < 8; ++i) {
    int k = ((i & 4) << 2) + kb + ((i & 3) << 1);
    a[2 * i]     = (__bf16)p[k];
    a[2 * i + 1] = (__bf16)p[k + 1];
  }
  return a;
}

// B fragment 32x16 bf16 for X @ W.T: Bmat[k][n] = W[nrow][k].
// Lane L: n = L&15 -> one row of W; lanes 0-15 read K {0..7,16..23}, lanes 16-31 {8..15,24..31}.
__device__ __forceinline__ v16bf load_b_rowptr(const __bf16* rowptr, int k0,
                                               int lane) {
  const __bf16* p = rowptr + k0 + ((lane >> 4) << 3);
  v16bf b;
#pragma unroll
  for (int i = 0; i < 16; ++i) b[i] = p[(i < 8) ? i : i + 8];
  return b;
}

__device__ __forceinline__ v16bf load_b_rows(const __bf16* W, int ldw, int n0,
                                             int k0, int lane) {
  return load_b_rowptr(W + (size_t)(n0 + (lane & 15)) * ldw, k0, lane);
}

// ---------------- init: bf16 weight conversion + state init ----------------
__global__ void k_init(const float* __restrict__ We, const float* __restrict__ Wh,
                       const float* __restrict__ Wa, const float* __restrict__ Ws,
                       const float* __restrict__ Wr, const float* __restrict__ mb,
                       unsigned char* __restrict__ ws) {
  size_t o = (size_t)blockIdx.x * blockDim.x + threadIdx.x;
  if (o < 131072) { ((__bf16*)(ws + OFF_WE))[o] = (__bf16)We[o]; return; }
  o -= 131072;
  if (o < 262144) { ((__bf16*)(ws + OFF_WH))[o] = (__bf16)Wh[o]; return; }
  o -= 262144;
  if (o < 262144) { ((__bf16*)(ws + OFF_WR))[o] = (__bf16)Wr[o]; return; }
  o -= 262144;
  if (o < 65536)  { ((__bf16*)(ws + OFF_WS))[o] = (__bf16)Ws[o]; return; }
  o -= 65536;
  if (o < 8192)   { ((__bf16*)(ws + OFF_WA))[o] = (__bf16)((o < 3072) ? Wa[o] : 0.f); return; }
  o -= 8192;
  if (o < 262144) {  // stack buffer 0 <- broadcast mem_bias
    int d = (int)(o & 63);
    int s = (int)((o >> 11) & 1);
    ((float*)(ws + OFF_STK))[o] = mb[s * 64 + d];
    return;
  }
  o -= 262144;
  if (o < 32768) { ((__bf16*)(ws + OFF_HID))[o] = (__bf16)0.f; return; }
  o -= 32768;
  if (o == 0) { *((unsigned*)(ws + OFF_CNT)) = 0u; }
}

// ---------------- pre = bf16(embs) @ We.T + be  (big parallel WMMA GEMM) ----
__global__ void __launch_bounds__(256) k_pre(const float* __restrict__ embs,
                                             const float* __restrict__ be,
                                             unsigned char* __restrict__ ws) {
  const __bf16* WeB = (const __bf16*)(ws + OFF_WE);
  __bf16* pre = (__bf16*)(ws + OFF_PRE);
  int lane = threadIdx.x & 31;
  int gw = blockIdx.x * 8 + (threadIdx.x >> 5);  // global wave id, 16384 total
  int m0 = (gw >> 3) * 16;                       // row tile of [32768]
  int n0 = (gw & 7) * 64;                        // 4 column tiles of HDIM

  v8f acc[4];
#pragma unroll
  for (int nt = 0; nt < 4; ++nt) { v8f z = {}; acc[nt] = z; }

  for (int kt = 0; kt < 8; ++kt) {               // K = 256
    v16bf a = load_a_f32(embs, IDIM_, m0, kt * 32, lane);
#pragma unroll
    for (int nt = 0; nt < 4; ++nt) {
      v16bf b = load_b_rows(WeB, IDIM_, n0 + nt * 16, kt * 32, lane);
      acc[nt] = wmma_bf16(a, b, acc[nt]);
    }
  }
  int hiRow = (lane >> 4) << 3;
#pragma unroll
  for (int nt = 0; nt < 4; ++nt) {
    int col = n0 + nt * 16 + (lane & 15);
    float bias = be[col];
#pragma unroll
    for (int i = 0; i < 8; ++i) {
      int row = m0 + hiRow + i;
      pre[(size_t)row * HDIM_ + col] = (__bf16)(acc[nt][i] + bias);
    }
  }
}

// ---------------- persistent recurrent kernel (8 WGs, 1 barrier/step) ------
// All WMMA operands staged in the WGP's 320KB LDS:
//   weights (160KB, loaded once) + hid stage (64KB/step) + tops (32KB/step).
__global__ void __launch_bounds__(256) k_rec(const int* __restrict__ lens,
                                             const float* __restrict__ bh,
                                             const float* __restrict__ ba,
                                             const float* __restrict__ bs,
                                             const float* __restrict__ br,
                                             const float* __restrict__ mb,
                                             unsigned char* __restrict__ ws,
                                             float* __restrict__ out) {
  __shared__ __align__(16) __bf16 WhS[64 * 512];    // this WG's 64 rows of Wh
  __shared__ __align__(16) __bf16 WrS[128 * 256];   // rows s*512+wg*64+n, s=0,1
  __shared__ __align__(16) __bf16 WsS[16 * 512];    // rows s*64 + wg*8 + d
  __shared__ __align__(16) __bf16 WaS[16 * 512];    // all 6 action rows (padded)
  __shared__ __align__(16) __bf16 hidS[64 * 512];   // hid_{t-1} stage
  __shared__ __align__(16) __bf16 topsS[64 * 256];  // [b][s*128 + cell*64 + d]
  __shared__ __align__(16) float  logitsS[64 * 8];  // [b][s*3+act], stride 8
  __shared__ __align__(16) float  pushS[64 * 16];   // [b][s*8 + d_local]

  const __bf16* WhB = (const __bf16*)(ws + OFF_WH);
  const __bf16* WrB = (const __bf16*)(ws + OFF_WR);
  const __bf16* WsB = (const __bf16*)(ws + OFF_WS);
  const __bf16* WaB = (const __bf16*)(ws + OFF_WA);
  const __bf16* pre = (const __bf16*)(ws + OFF_PRE);
  __bf16* hidB = (__bf16*)(ws + OFF_HID);
  float*  stk  = (float*)(ws + OFF_STK);
  unsigned* cnt = (unsigned*)(ws + OFF_CNT);

  const int wg = blockIdx.x;             // 0..7: owns HDIM cols [wg*64, wg*64+64)
  const int tid = threadIdx.x;           //            and SDIM dims [wg*8, wg*8+8)
  const int lane = tid & 31;
  const int wv = tid >> 5;               // 8 waves
  const int hiRow = (lane >> 4) << 3;

  // ---- one-time weight staging into LDS ----
  for (int e = tid; e < 64 * 512; e += 256)
    WhS[e] = WhB[(size_t)(wg * 64 + (e >> 9)) * 512 + (e & 511)];
  for (int e = tid; e < 128 * 256; e += 256) {
    int r = e >> 8;
    WrS[e] = WrB[(size_t)((r >> 6) * 512 + wg * 64 + (r & 63)) * 256 + (e & 255)];
  }
  for (int e = tid; e < 16 * 512; e += 256) {
    int r = e >> 9;
    int wsrow = (r < 8) ? (wg * 8 + r) : (64 + wg * 8 + (r - 8));
    WsS[e] = WsB[(size_t)wsrow * 512 + (e & 511)];
  }
  for (int e = tid; e < 16 * 512; e += 256) WaS[e] = WaB[e];

  float* outHidLast = out + (size_t)T_ * B_ * HDIM_;
  float* outRes = outHidLast + (size_t)B_ * HDIM_;

  int p = 0;
  for (int t = 0; t < T_; ++t) {
    const __bf16* hidP = hidB + p * (B_ * HDIM_);
    __bf16* hidN = hidB + (1 - p) * (B_ * HDIM_);
    const float* stkP = stk + (size_t)p * 262144;
    float* stkN = stk + (size_t)(1 - p) * 262144;

    // Phase A: stage hid (vectorized) and bf16 tops tile; prefetch next pre.
    {
      const uint4* hg = (const uint4*)hidP;
      uint4* hs = (uint4*)hidS;
      for (int e = tid; e < (64 * 512) / 8; e += 256) hs[e] = hg[e];
    }
    for (int e = tid; e < B_ * 256; e += 256) {
      int b = e >> 8, k = e & 255;
      int s = k >> 7, c = (k >> 6) & 1, d = k & 63;
      topsS[e] = (__bf16)stkP[(size_t)(((b * 2 + s) * 32) + c) * 64 + d];
    }
    if (t + 1 < T_ && tid < 64)
      __builtin_prefetch(pre + ((size_t)(t + 1) * B_ + tid) * HDIM_, 0, 0);
    __syncthreads();

    // Phase B1: mhid tiles. Wave wv: Mtile = wv&3, Ntiles {wv>>2, (wv>>2)+2}.
    {
      int mt = wv & 3, ntA = wv >> 2;
      int m0 = mt * 16;
      v8f accs[2];
#pragma unroll
      for (int j = 0; j < 2; ++j) { v8f z = {}; accs[j] = z; }

      for (int kt = 0; kt < 16; ++kt) {                    // hid @ Wh.T
        v16bf a = load_a_bf16(hidS, HDIM_, m0, kt * 32, lane);
#pragma unroll
        for (int j = 0; j < 2; ++j) {
          v16bf b = load_b_rows(WhS, HDIM_, (ntA + 2 * j) * 16, kt * 32, lane);
          accs[j] = wmma_bf16(a, b, accs[j]);
        }
      }
      for (int s = 0; s < 2; ++s)                          // read: tops @ Wr.T
        for (int kt = 0; kt < 8; ++kt) {
          v16bf a = load_a_bf16(topsS, 256, m0, kt * 32, lane);
#pragma unroll
          for (int j = 0; j < 2; ++j) {
            v16bf b = load_b_rows(WrS, 256, s * 64 + (ntA + 2 * j) * 16,
                                  kt * 32, lane);
            accs[j] = wmma_bf16(a, b, accs[j]);
          }
        }
#pragma unroll
      for (int j = 0; j < 2; ++j) {
        int col = wg * 64 + (ntA + 2 * j) * 16 + (lane & 15);
        float bias = bh[col] + br[col] + br[512 + col];
#pragma unroll
        for (int i = 0; i < 8; ++i) {
          int row = m0 + hiRow + i;                        // batch index
          float v = accs[j][i] + bias +
                    (float)pre[((size_t)t * B_ + row) * HDIM_ + col];
          v = tanhf(v);
          out[((size_t)t * B_ + row) * HDIM_ + col] = v;
          hidN[row * HDIM_ + col] = (__bf16)v;
          if (t == lens[row] - 1) outHidLast[row * HDIM_ + col] = v;
        }
      }
    }

    // Phase B2: action logits (waves 0-3, redundant per WG) / push vals (4-7).
    if (wv < 4) {
      int m0 = wv * 16;
      v8f acc = {};
      for (int kt = 0; kt < 16; ++kt) {
        v16bf a = load_a_bf16(hidS, HDIM_, m0, kt * 32, lane);
        v16bf b = load_b_rows(WaS, HDIM_, 0, kt * 32, lane);
        acc = wmma_bf16(a, b, acc);
      }
      int col = lane & 15;                                 // n = s*3 + act
      if (col < 6) {
        float bias = ba[col];
#pragma unroll
        for (int i = 0; i < 8; ++i)
          logitsS[(m0 + hiRow + i) * 8 + col] = acc[i] + bias;
      }
    } else {
      int m0 = (wv - 4) * 16;
      v8f acc = {};
      int n = lane & 15;                                   // local col 0..15
      int wsrow = (n < 8) ? (wg * 8 + n) : (64 + wg * 8 + (n - 8));  // s*64+d
      const __bf16* rowp = WsS + (size_t)n * 512;
      for (int kt = 0; kt < 16; ++kt) {
        v16bf a = load_a_bf16(hidS, HDIM_, m0, kt * 32, lane);
        v16bf b = load_b_rowptr(rowp, kt * 32, lane);
        acc = wmma_bf16(a, b, acc);
      }
      float bias = bs[wsrow];
#pragma unroll
      for (int i = 0; i < 8; ++i)
        pushS[(m0 + hiRow + i) * 16 + n] = tanhf(acc[i] + bias);
    }
    __syncthreads();

    // Phase C: stack blend for this WG's SDIM slice (d = wg*8 .. wg*8+7).
    for (int e = tid; e < 32768; e += 256) {
      int dl = e & 7, c = (e >> 3) & 31, s = (e >> 8) & 1, b = e >> 9;
      int d = wg * 8 + dl;
      float l0 = logitsS[b * 8 + s * 3 + 0];
      float l1 = logitsS[b * 8 + s * 3 + 1];
      float l2 = logitsS[b * 8 + s * 3 + 2];
      float mx = fmaxf(l0, fmaxf(l1, l2));
      float e0 = __expf(l0 - mx), e1 = __expf(l1 - mx), e2 = __expf(l2 - mx);
      float inv = 1.f / (e0 + e1 + e2);
      size_t base = (size_t)((b * 2 + s) * 32) * 64 + d;
      float cur = stkP[base + (size_t)c * 64];
      float up = (c == 0) ? pushS[b * 16 + s * 8 + dl]
                          : stkP[base + (size_t)(c - 1) * 64];
      float dn = (c == SSZ_ - 1) ? mb[s * 64 + d]
                                 : stkP[base + (size_t)(c + 1) * 64];
      float nv = e0 * inv * up + e1 * inv * dn + e2 * inv * cur;
      stkN[base + (size_t)c * 64] = nv;
      if (t == lens[b] - 1) outRes[base + (size_t)c * 64] = nv;
    }

    // Device-wide barrier (monotonic counter; counter zeroed by k_init).
    __syncthreads();
    if (tid == 0) {
      __threadfence();
      atomicAdd(cnt, 1u);
      unsigned target = (unsigned)NWG_ * (unsigned)(t + 1);
      while (__hip_atomic_load(cnt, __ATOMIC_RELAXED,
                               __HIP_MEMORY_SCOPE_AGENT) < target)
        __builtin_amdgcn_s_sleep(2);
    }
    __syncthreads();
    __threadfence();
    p ^= 1;
  }
}

extern "C" void kernel_launch(void* const* d_in, const int* in_sizes, int n_in,
                              void* d_out, int out_size, void* d_ws, size_t ws_size,
                              hipStream_t stream) {
  const float* embs = (const float*)d_in[0];
  const int*   lens = (const int*)d_in[1];
  const float* We = (const float*)d_in[2];
  const float* be = (const float*)d_in[3];
  const float* Wh = (const float*)d_in[4];
  const float* bh = (const float*)d_in[5];
  const float* Wa = (const float*)d_in[6];
  const float* ba = (const float*)d_in[7];
  const float* Ws = (const float*)d_in[8];
  const float* bs = (const float*)d_in[9];
  const float* Wr = (const float*)d_in[10];
  const float* br = (const float*)d_in[11];
  const float* mb = (const float*)d_in[12];
  unsigned char* ws = (unsigned char*)d_ws;
  float* out = (float*)d_out;

  (void)in_sizes; (void)n_in; (void)out_size; (void)ws_size;

  k_init<<<(1024001 + 255) / 256, 256, 0, stream>>>(We, Wh, Wa, Ws, Wr, mb, ws);
  k_pre<<<2048, 256, 0, stream>>>(embs, be, ws);
  k_rec<<<NWG_, 256, 0, stream>>>(lens, bh, ba, bs, br, mb, ws, out);
}
